// MessagePassingConvolution_61641370632667
// MI455X (gfx1250) — matrix operand order repaired and checked by reading, not compiled, not run on hardware
//
#include <hip/hip_runtime.h>
#include <hip/hip_bf16.h>
#include <math.h>

typedef __attribute__((ext_vector_type(16))) _Float16 v16h;
typedef __attribute__((ext_vector_type(8)))  _Float16 v8h;
typedef __attribute__((ext_vector_type(8)))  float    v8f;

#define INV_SQRT3  0.57735026918962576f
#define INV_SQRT20 0.22360679774997896f
#define INV_SQRT32 0.17677669529663687f

static __device__ __forceinline__ v8f wmma_f16(v16h a, v16h b, v8f c) {
  // 8 args: (neg_a, A, neg_b, B, c_mod, C, reuse_a, reuse_b)
  return __builtin_amdgcn_wmma_f32_16x16x32_f16(false, a, false, b, (short)0, c, false, false);
}

// fast silu: x * rcp(1+exp(-x)) -> v_exp_f32 + v_add + v_rcp_f32 + v_mul
static __device__ __forceinline__ float silu(float x) {
  return x * __builtin_amdgcn_rcpf(1.0f + __expf(-x));
}

static __device__ __forceinline__ v16h pack16(v8h lo, v8h hi) {
  v16h r;
#pragma unroll
  for (int i = 0; i < 8; ++i) { r[i] = lo[i]; r[i + 8] = hi[i]; }
  return r;
}

__global__ void zero_kernel(float* __restrict__ out, int n) {
  int i = blockIdx.x * blockDim.x + threadIdx.x;
  if (i < n) out[i] = 0.0f;
}

__global__ void __launch_bounds__(256)
mpconv_kernel(const float* __restrict__ edge_feats,   // (E,128)
              const float* __restrict__ edge_attrs,   // (E,4)
              const int*   __restrict__ receivers,    // (E,)
              const float* __restrict__ W1,           // (32,64)
              const float* __restrict__ W2,           // (64,64)
              const float* __restrict__ W3,           // (64,128)
              float* __restrict__ out,                // (N,256), pre-zeroed
              int E)
{
  // f16 weights, transposed [n][k] with 1/sqrt(K) folded in
  __shared__ __align__(32) _Float16 w1t[64 * 32];
  __shared__ __align__(32) _Float16 w2t[64 * 64];
  __shared__ __align__(32) _Float16 w3t[128 * 64];
  // per-wave activation staging (D-layout -> A-layout re-shuffle)
  __shared__ __align__(32) _Float16 hb1[8][16 * 64];
  __shared__ __align__(32) _Float16 hb2[8][16 * 64];

  const int tid  = threadIdx.x;
  const int lane = tid & 31;
  const int wave = tid >> 5;
  const int cl   = lane & 15;   // N-column within 16x16 tile
  const int hf   = lane >> 4;   // lane half (K/M split per ISA layout)

  // ---- convert weights -> LDS (f16, transposed, prescaled) ----
  for (int i = tid; i < 64 * 32; i += 256) {
    int n = i >> 5, k = i & 31;
    w1t[i] = (_Float16)(W1[k * 64 + n] * INV_SQRT32);
  }
  for (int i = tid; i < 64 * 64; i += 256) {
    int n = i >> 6, k = i & 63;
    w2t[i] = (_Float16)(W2[k * 64 + n] * 0.125f);
  }
  for (int i = tid; i < 128 * 64; i += 256) {
    int n = i >> 6, k = i & 63;
    w3t[i] = (_Float16)(W3[k * 128 + n] * 0.125f);
  }
  __syncthreads();

  const long long e0 = (long long)blockIdx.x * 128 + (long long)wave * 16;

  // ---- A matrix for layer 1: s tile (16 edges x 32 ch), contiguous K order (k = 16*hf + s)
  long long erow = e0 + cl; if (erow >= E) erow = (long long)E - 1;
  const float* sp = edge_feats + erow * 128;
  v16h a_s;
#pragma unroll
  for (int i = 0; i < 8; ++i) a_s[i]     = (_Float16)sp[16 * hf + i];       // K=16*hf+0..7
#pragma unroll
  for (int i = 0; i < 8; ++i) a_s[i + 8] = (_Float16)sp[16 * hf + 8 + i];   // K=16*hf+8..15

  _Float16* hb1w = hb1[wave];
  _Float16* hb2w = hb2[wave];

  // ---- layer 1: (16x32)@(32x64), 4 WMMAs ----
#pragma unroll
  for (int t = 0; t < 4; ++t) {
    v16h b = *(const v16h*)&w1t[(16 * t + cl) * 32 + 16 * hf];
    v8f acc = {};
    acc = wmma_f16(a_s, b, acc);
#pragma unroll
    for (int r = 0; r < 8; ++r)
      hb1w[(r + 8 * hf) * 64 + 16 * t + cl] = (_Float16)silu(acc[r]);
  }
  __syncthreads();

  // reload h1 in A layout (16x64 as two K=32 chunks)
  const _Float16* hp1 = &hb1w[cl * 64];
  v16h a1lo = pack16(*(const v8h*)(hp1 + 16 * hf),      *(const v8h*)(hp1 + 8 + 16 * hf));
  v16h a1hi = pack16(*(const v8h*)(hp1 + 32 + 16 * hf), *(const v8h*)(hp1 + 40 + 16 * hf));

  // ---- layer 2: (16x64)@(64x64), 8 WMMAs ----
#pragma unroll
  for (int t = 0; t < 4; ++t) {
    const _Float16* bp = &w2t[(16 * t + cl) * 64];
    v16h blo = *(const v16h*)(bp + 16 * hf);
    v16h bhi = *(const v16h*)(bp + 32 + 16 * hf);
    v8f acc = {};
    acc = wmma_f16(a1lo, blo, acc);
    acc = wmma_f16(a1hi, bhi, acc);
#pragma unroll
    for (int r = 0; r < 8; ++r)
      hb2w[(r + 8 * hf) * 64 + 16 * t + cl] = (_Float16)silu(acc[r]);
  }
  __syncthreads();

  const _Float16* hp2 = &hb2w[cl * 64];
  v16h a2lo = pack16(*(const v8h*)(hp2 + 16 * hf),      *(const v8h*)(hp2 + 8 + 16 * hf));
  v16h a2hi = pack16(*(const v8h*)(hp2 + 32 + 16 * hf), *(const v8h*)(hp2 + 40 + 16 * hf));

  // ---- layer 3: all 8 N-tiles (16 WMMAs) ----
  v8f acc[8];
#pragma unroll
  for (int t = 0; t < 8; ++t) {
    const _Float16* bp = &w3t[(16 * t + cl) * 64];
    v16h blo = *(const v16h*)(bp + 16 * hf);
    v16h bhi = *(const v16h*)(bp + 32 + 16 * hf);
    v8f a = {};
    a = wmma_f16(a2lo, blo, a);
    acc[t] = wmma_f16(a2hi, bhi, a);
  }

  // ---- messages + scatter-add: rows outer (hoist per-edge scalars) ----
#pragma unroll
  for (int r = 0; r < 8; ++r) {
    const int m = r + 8 * hf;                   // edge row within tile (D layout)
    const long long e  = e0 + m;
    const long long eg = (e < E) ? e : (long long)E - 1;
    const bool ok = (e < E);
    const float* fp = edge_feats + eg * 128;
    const float  s0  = edge_attrs[eg * 4 + 0];
    const float  v0x = edge_attrs[eg * 4 + 1];
    const float  v0y = edge_attrs[eg * 4 + 2];
    const float  v0z = edge_attrs[eg * 4 + 3];
    float* op = out + (size_t)receivers[eg] * 256;

#pragma unroll
    for (int t = 0; t < 8; ++t) {
      const float mix = acc[t][r] * INV_SQRT20; // fold final 1/sqrt(20)
      const int c = 16 * t + cl;                // mix channel 0..127
      if (t < 2) {                              // msg_s[:, c] = s*s0*mix
        float val = fp[c] * s0 * mix;
        if (ok) atomicAdd(op + c, val);
      } else if (t < 4) {                       // msg_s[:, 32+j] = (v.v0)/sqrt3*mix
        const int j = c - 32;
        float d = fp[32 + 3 * j] * v0x + fp[33 + 3 * j] * v0y + fp[34 + 3 * j] * v0z;
        if (ok) atomicAdd(op + c, d * INV_SQRT3 * mix);
      } else if (t < 6) {                       // msg_v[j][d] = s[j]*v0[d]*mix
        const int j = c - 64;
        float sv = fp[j] * mix;
        if (ok) {
          atomicAdd(op + 64 + 3 * j + 0, sv * v0x);
          atomicAdd(op + 64 + 3 * j + 1, sv * v0y);
          atomicAdd(op + 64 + 3 * j + 2, sv * v0z);
        }
      } else {                                  // msg_v[32+j][d] = v[j][d]*s0*mix
        const int j = c - 96;
        float s0m = s0 * mix;
        if (ok) {
          atomicAdd(op + 64 + 3 * (32 + j) + 0, fp[32 + 3 * j + 0] * s0m);
          atomicAdd(op + 64 + 3 * (32 + j) + 1, fp[32 + 3 * j + 1] * s0m);
          atomicAdd(op + 64 + 3 * (32 + j) + 2, fp[32 + 3 * j + 2] * s0m);
        }
      }
    }
  }
}

extern "C" void kernel_launch(void* const* d_in, const int* in_sizes, int n_in,
                              void* d_out, int out_size, void* d_ws, size_t ws_size,
                              hipStream_t stream) {
  const float* edge_feats = (const float*)d_in[0];
  const float* edge_attrs = (const float*)d_in[1];
  const int*   receivers  = (const int*)d_in[2];
  // d_in[3] = n_nodes scalar (device); use out_size/256 instead on host
  const float* W1 = (const float*)d_in[4];
  const float* W2 = (const float*)d_in[5];
  const float* W3 = (const float*)d_in[6];
  float* out = (float*)d_out;

  const int E = in_sizes[0] / 128;

  zero_kernel<<<(out_size + 255) / 256, 256, 0, stream>>>(out, out_size);

  const int blocks = (E + 127) / 128;  // 128 edges per block (8 waves x 16)
  mpconv_kernel<<<blocks, 256, 0, stream>>>(edge_feats, edge_attrs, receivers,
                                            W1, W2, W3, out, E);
}
